// TransformerBlock_53798760350454
// MI455X (gfx1250) — compile-verified, matrix operand order checked by reading, not run on hardware
//
#include <hip/hip_runtime.h>

// ---------------------------------------------------------------------------
// Feature detection: CDNA5 async global->LDS copies (ASYNCcnt-tracked)
// ---------------------------------------------------------------------------
#if defined(__has_builtin)
#  if __has_builtin(__builtin_amdgcn_global_load_async_to_lds_b128) && \
      __has_builtin(__builtin_amdgcn_s_wait_asynccnt)
#    define HAVE_ASYNC 1
#  endif
#endif
#ifndef HAVE_ASYNC
#  define HAVE_ASYNC 0
#endif

// Builtin params are int4 pointers: arg0 = addrspace(1) global, arg1 = LDS.
typedef int v4i_vs __attribute__((vector_size(16)));
#define AS1CAST(p) ((__attribute__((address_space(1))) v4i_vs*)(p))
#define AS3CAST(p) ((__attribute__((address_space(3))) v4i_vs*)(p))

// ---------------------------------------------------------------------------
// Types / helpers
// ---------------------------------------------------------------------------
typedef unsigned short u16t;
typedef unsigned int   u32t;
typedef __attribute__((ext_vector_type(16))) __bf16 v16bf;
typedef __attribute__((ext_vector_type(8)))  float  v8f;

union Frag16 {
    v16bf bf;
    u32t  u32[8];
    u16t  u16[16];
};

__device__ __forceinline__ u16t f2bf(float f) {
    u32t u = __float_as_uint(f);
    u32t r = u + 0x7FFFu + ((u >> 16) & 1u);   // round-to-nearest-even
    return (u16t)(r >> 16);
}

// K index of the first element of VGPR pair r for the 16-bit WMMA A/B layout.
// lane<16 (hsel=0): K = {0,2,4,6,16,18,20,22}; lane>=16 (hsel=1): +8.
__device__ __forceinline__ int kmap(int r, int hsel) {
    return (r < 4 ? 2 * r : 2 * r + 8) + hsel * 8;
}

__device__ __forceinline__ float gelu_tanh(float x) {
    const float c = 0.7978845608028654f; // sqrt(2/pi)
    return 0.5f * x * (1.0f + tanhf(c * (x + 0.044715f * x * x * x)));
}

// ---------------------------------------------------------------------------
// f32 -> bf16 transposed conversion: dst[n*K + k] = bf16(src[k*N + n]).
// 32x32 tiles through LDS; grid = (N/32, K/32), block = 256.
// ---------------------------------------------------------------------------
__global__ __launch_bounds__(256) void conv_bf16_t(
    const float* __restrict__ src, u16t* __restrict__ dst, int K, int N)
{
    __shared__ float t[32][33];
    const int n0 = blockIdx.x * 32, k0 = blockIdx.y * 32;
    const int c = threadIdx.x & 31, r = threadIdx.x >> 5;   // r = 0..7
    #pragma unroll
    for (int i = 0; i < 4; ++i)
        t[r + 8 * i][c] = src[(size_t)(k0 + r + 8 * i) * N + n0 + c];
    __syncthreads();
    #pragma unroll
    for (int i = 0; i < 4; ++i)
        dst[(size_t)(n0 + r + 8 * i) * K + k0 + c] = f2bf(t[c][r + 8 * i]);
}

// ---------------------------------------------------------------------------
// LayerNorm (biased variance, eps=1e-5) -> bf16 activations; one block / row
// ---------------------------------------------------------------------------
__global__ __launch_bounds__(256) void layernorm_bf16(
    const float* __restrict__ x, const float* __restrict__ g,
    const float* __restrict__ s, u16t* __restrict__ out, int Dn)
{
    __shared__ float red[256];
    const int row = blockIdx.x;
    const int tid = threadIdx.x;
    const float* xr = x + (size_t)row * Dn;

    float lsum = 0.f;
    for (int i = tid; i < Dn; i += 256) lsum += xr[i];
    red[tid] = lsum; __syncthreads();
    for (int st = 128; st > 0; st >>= 1) { if (tid < st) red[tid] += red[tid + st]; __syncthreads(); }
    float mean = red[0] / (float)Dn;
    __syncthreads();

    float lvar = 0.f;
    for (int i = tid; i < Dn; i += 256) { float d = xr[i] - mean; lvar += d * d; }
    red[tid] = lvar; __syncthreads();
    for (int st = 128; st > 0; st >>= 1) { if (tid < st) red[tid] += red[tid + st]; __syncthreads(); }
    float rstd = rsqrtf(red[0] / (float)Dn + 1e-5f);

    u16t* orow = out + (size_t)row * Dn;
    for (int i = tid; i < Dn; i += 256)
        orow[i] = f2bf((xr[i] - mean) * rstd * g[i] + s[i]);
}

// ---------------------------------------------------------------------------
// Tile staging. Both operands are K-contiguous 128x32 tiles (A row-major,
// B stored pre-transposed [N][K]); pitch 40 u16 keeps 16B chunks aligned and
// fragment gathers bank-conflict-free. 4 x 16B copies per thread per stage.
// ---------------------------------------------------------------------------
#define T_PITCH 40

__device__ __forceinline__ void stage_tiles(
    const u16t* __restrict__ Ag, const u16t* __restrict__ BTg,
    u16t* dA, u16t* dB, int bm, int bn, int K, int k0, int tid)
{
    const int tr = tid >> 2, ts = tid & 3;   // 128 rows x 4 chunks
#if HAVE_ASYNC
    #pragma unroll
    for (int it = 0; it < 2; ++it) {
        const u16t* src = Ag + (size_t)(bm + tr + it * 64) * K + k0 + ts * 8;
        u16t*       dst = dA + (tr + it * 64) * T_PITCH + ts * 8;
        __builtin_amdgcn_global_load_async_to_lds_b128(AS1CAST(src), AS3CAST(dst), 0, 0);
    }
    #pragma unroll
    for (int it = 0; it < 2; ++it) {
        const u16t* src = BTg + (size_t)(bn + tr + it * 64) * K + k0 + ts * 8;
        u16t*       dst = dB + (tr + it * 64) * T_PITCH + ts * 8;
        __builtin_amdgcn_global_load_async_to_lds_b128(AS1CAST(src), AS3CAST(dst), 0, 0);
    }
#else
    #pragma unroll
    for (int it = 0; it < 2; ++it) {
        uint4 v = *(const uint4*)(Ag + (size_t)(bm + tr + it * 64) * K + k0 + ts * 8);
        *(uint4*)(dA + (tr + it * 64) * T_PITCH + ts * 8) = v;
    }
    #pragma unroll
    for (int it = 0; it < 2; ++it) {
        uint4 v = *(const uint4*)(BTg + (size_t)(bn + tr + it * 64) * K + k0 + ts * 8);
        *(uint4*)(dB + (tr + it * 64) * T_PITCH + ts * 8) = v;
    }
#endif
}

// ---------------------------------------------------------------------------
// bf16 GEMM: C[M,N] = A[M,K] @ B[K,N] with B given TRANSPOSED ([N][K] bf16).
// (+bias) (gelu?) (+resid) -> f32 / bf16 / bf16-transposed outputs.
// 128x128 block tile, 8 waves (2x4), wave tile 64x32, K step 32,
// double-buffered LDS with async copy pipelining.
// ---------------------------------------------------------------------------
__global__ __launch_bounds__(256) void gemm_bf16(
    const u16t* __restrict__ Ag, const u16t* __restrict__ BTg,
    const float* __restrict__ bias, const float* __restrict__ resid,
    float* __restrict__ outF, u16t* __restrict__ outB, u16t* __restrict__ outT,
    int M, int N, int K, int doGelu)
{
    __shared__ __align__(16) u16t ldsA[2][128 * T_PITCH];
    __shared__ __align__(16) u16t ldsB[2][128 * T_PITCH];

    const int tid  = threadIdx.x;
    const int lane = tid & 31;
    const int wave = tid >> 5;
    const int lr   = lane & 15;
    const int hsel = lane >> 4;
    const int wm   = (wave >> 2) * 64;
    const int wn   = (wave & 3) * 32;
    const int bm   = blockIdx.y * 128;
    const int bn   = blockIdx.x * 128;

    v8f acc[4][2] = {};

    stage_tiles(Ag, BTg, ldsA[0], ldsB[0], bm, bn, K, 0, tid);

    int buf = 0;
    for (int k0 = 0; k0 < K; k0 += 32) {
        const int nk = k0 + 32;
        if (nk < K)
            stage_tiles(Ag, BTg, ldsA[buf ^ 1], ldsB[buf ^ 1], bm, bn, K, nk, tid);
#if HAVE_ASYNC
        // async loads retire in order: waiting to <=4 retires the 4 copies of
        // the CURRENT chunk while the next chunk's 4 stay in flight.
        if (nk < K) __builtin_amdgcn_s_wait_asynccnt(4);
        else        __builtin_amdgcn_s_wait_asynccnt(0);
#endif
        __syncthreads();

        const u32t* cA32 = (const u32t*)ldsA[buf];
        const u32t* cB32 = (const u32t*)ldsB[buf];

        Frag16 af[4], bfr[2];
        #pragma unroll
        for (int mf = 0; mf < 4; ++mf) {
            int row = wm + mf * 16 + lr;
            #pragma unroll
            for (int r = 0; r < 8; ++r)
                af[mf].u32[r] = cA32[(row * T_PITCH + kmap(r, hsel)) >> 1];
        }
        #pragma unroll
        for (int nf = 0; nf < 2; ++nf) {
            int col = wn + nf * 16 + lr;
            #pragma unroll
            for (int r = 0; r < 8; ++r)
                bfr[nf].u32[r] = cB32[(col * T_PITCH + kmap(r, hsel)) >> 1];
        }
        #pragma unroll
        for (int mf = 0; mf < 4; ++mf)
            #pragma unroll
            for (int nf = 0; nf < 2; ++nf)
                acc[mf][nf] = __builtin_amdgcn_wmma_f32_16x16x32_bf16(
                    false, af[mf].bf, false, bfr[nf].bf, (short)0, acc[mf][nf], false, false);

        __syncthreads();   // reads of this buffer done before it is re-staged
        buf ^= 1;
    }

    // Epilogue: bias -> gelu -> residual -> store
    #pragma unroll
    for (int mf = 0; mf < 4; ++mf) {
        #pragma unroll
        for (int nf = 0; nf < 2; ++nf) {
            #pragma unroll
            for (int r = 0; r < 8; ++r) {
                int rowg = bm + wm + mf * 16 + r + hsel * 8;
                int colg = bn + wn + nf * 16 + lr;
                float v = acc[mf][nf][r];
                if (bias)   v += bias[colg];
                if (doGelu) v = gelu_tanh(v);
                size_t oidx = (size_t)rowg * N + colg;
                if (resid) v += resid[oidx];
                if (outF)  outF[oidx] = v;
                if (outB)  outB[oidx] = f2bf(v);
                if (outT)  outT[(size_t)colg * M + rowg] = f2bf(v);
            }
        }
    }
}

// ---------------------------------------------------------------------------
// Flash attention: one wave per (b, h, 16-row q tile). Causal mask before
// scaling (scale = 1/sqrt(D) = 1/sqrt(2048), faithful to the reference quirk).
// V is consumed TRANSPOSED ([D][B*S] bf16) so P@V B-fragments gather aligned
// u32 pairs of consecutive keys.
// ---------------------------------------------------------------------------
#define S_LEN   2048
#define D_MODEL 2048
#define HD_     128
#define NH_     16
#define MS_TOT  4096

__global__ __launch_bounds__(32) void attention_flash(
    const u16t* __restrict__ Q, const u16t* __restrict__ Km,
    const u16t* __restrict__ VT, u16t* __restrict__ Octx)
{
    __shared__ __align__(16) float ldsP[16 * 32];
    const int lane = threadIdx.x;
    const int lr   = lane & 15;
    const int hsel = lane >> 4;
    const int ntile = S_LEN / 16;            // 128
    const int bh = blockIdx.x / ntile;
    const int qt = blockIdx.x % ntile;
    const int b  = bh / NH_;
    const int h  = bh % NH_;
    const int rowbase = b * S_LEN;
    const int hcol = h * HD_;
    const float scale = 0.02209708691207961f; // 1/sqrt(2048)

    const u32t* q32 = (const u32t*)Q;
    const u32t* k32 = (const u32t*)Km;
    const u32t* vt32 = (const u32t*)VT;

    // Q fragments: 16 rows x 128 dims -> 4 A-frags
    Frag16 aq[4];
    #pragma unroll
    for (int f = 0; f < 4; ++f) {
        size_t rq = (size_t)(rowbase + qt * 16 + lr) * D_MODEL + hcol + f * 32;
        #pragma unroll
        for (int r = 0; r < 8; ++r)
            aq[f].u32[r] = q32[(rq + kmap(r, hsel)) >> 1];
    }

    float mrow[8], lrow[8], alpha[8];
    #pragma unroll
    for (int r = 0; r < 8; ++r) { mrow[r] = -1e30f; lrow[r] = 0.f; }
    v8f ctx[8] = {};

    const int nsteps = (qt * 16 + 16 + 31) / 32;
    for (int st = 0; st < nsteps; ++st) {
        const int kb = st * 32;

        // prefetch next step's K rows / V columns (global_prefetch_b8)
        if (st + 1 < nsteps) {
            __builtin_prefetch(Km + (size_t)(rowbase + kb + 32 + lr) * D_MODEL + hcol, 0, 0);
            __builtin_prefetch(VT + (size_t)(hcol + lr) * MS_TOT + rowbase + kb + 32, 0, 0);
        }

        // scores: q(16x128) @ k^T -> two 16x16 tiles (keys kb..+15, kb+16..+31)
        v8f s0 = {}, s1 = {};
        #pragma unroll
        for (int f = 0; f < 4; ++f) {
            Frag16 bk0, bk1;
            size_t rk0 = (size_t)(rowbase + kb + lr) * D_MODEL + hcol + f * 32;
            size_t rk1 = (size_t)(rowbase + kb + 16 + lr) * D_MODEL + hcol + f * 32;
            #pragma unroll
            for (int r = 0; r < 8; ++r) {
                bk0.u32[r] = k32[(rk0 + kmap(r, hsel)) >> 1];
                bk1.u32[r] = k32[(rk1 + kmap(r, hsel)) >> 1];
            }
            s0 = __builtin_amdgcn_wmma_f32_16x16x32_bf16(false, aq[f].bf, false, bk0.bf, (short)0, s0, false, false);
            s1 = __builtin_amdgcn_wmma_f32_16x16x32_bf16(false, aq[f].bf, false, bk1.bf, (short)0, s1, false, false);
        }

        // online softmax update (rows are per-(vgpr,half); reduce across 16 lanes)
        #pragma unroll
        for (int r = 0; r < 8; ++r) {
            int qrow = qt * 16 + r + hsel * 8;
            int key0 = kb + lr, key1 = kb + 16 + lr;
            float v0 = (key0 > qrow) ? -1e30f : s0[r] * scale;
            float v1 = (key1 > qrow) ? -1e30f : s1[r] * scale;
            float mx = fmaxf(v0, v1);
            #pragma unroll
            for (int msk = 8; msk >= 1; msk >>= 1) mx = fmaxf(mx, __shfl_xor(mx, msk, 32));
            float mnew = fmaxf(mrow[r], mx);
            alpha[r] = __expf(mrow[r] - mnew);
            float p0 = __expf(v0 - mnew);
            float p1 = __expf(v1 - mnew);
            float psum = p0 + p1;
            #pragma unroll
            for (int msk = 8; msk >= 1; msk >>= 1) psum += __shfl_xor(psum, msk, 32);
            lrow[r] = lrow[r] * alpha[r] + psum;
            mrow[r] = mnew;
            ldsP[(r + hsel * 8) * 32 + lr]      = p0;
            ldsP[(r + hsel * 8) * 32 + 16 + lr] = p1;
        }
        __syncthreads();

        // transpose P (16x32) into WMMA A-fragment layout, f32 -> bf16
        Frag16 pa;
        #pragma unroll
        for (int r = 0; r < 8; ++r) {
            int kk = kmap(r, hsel);
            u32t lo = f2bf(ldsP[lr * 32 + kk]);
            u32t hi = f2bf(ldsP[lr * 32 + kk + 1]);
            pa.u32[r] = lo | (hi << 16);
        }
        __syncthreads();

        // ctx = ctx*alpha + P(16x32) @ V(32x128); V^T gathers are u32 pairs
        #pragma unroll
        for (int nf = 0; nf < 8; ++nf) {
            Frag16 bv;
            #pragma unroll
            for (int r = 0; r < 8; ++r) {
                size_t vi = (size_t)(hcol + nf * 16 + lr) * MS_TOT
                          + rowbase + kb + kmap(r, hsel);
                bv.u32[r] = vt32[vi >> 1];
            }
            v8f c = ctx[nf];
            #pragma unroll
            for (int r = 0; r < 8; ++r) c[r] *= alpha[r];
            ctx[nf] = __builtin_amdgcn_wmma_f32_16x16x32_bf16(false, pa.bf, false, bv.bf, (short)0, c, false, false);
        }
    }

    // normalize and store ctx as bf16 in [B*S, D] layout (col = h*128 + d)
    #pragma unroll
    for (int nf = 0; nf < 8; ++nf) {
        #pragma unroll
        for (int r = 0; r < 8; ++r) {
            float v = ctx[nf][r] / lrow[r];
            size_t oidx = (size_t)(rowbase + qt * 16 + r + hsel * 8) * D_MODEL
                        + hcol + nf * 16 + lr;
            Octx[oidx] = f2bf(v);
        }
    }
}

// ---------------------------------------------------------------------------
// Orchestration
// ---------------------------------------------------------------------------
extern "C" void kernel_launch(void* const* d_in, const int* in_sizes, int n_in,
                              void* d_out, int out_size, void* d_ws, size_t ws_size,
                              hipStream_t stream) {
    (void)in_sizes; (void)n_in; (void)out_size; (void)ws_size;
    const int Bn = 2, Sn = 2048, Dn = 2048, FF = 8192;
    const int MS = Bn * Sn; // 4096 rows

    const float* x  = (const float*)d_in[0];
    const float* Wq = (const float*)d_in[1];
    const float* Wk = (const float*)d_in[2];
    const float* Wv = (const float*)d_in[3];
    const float* Wo = (const float*)d_in[4];
    const float* bo = (const float*)d_in[5];
    const float* W1 = (const float*)d_in[6];
    const float* b1 = (const float*)d_in[7];
    const float* W2 = (const float*)d_in[8];
    const float* b2 = (const float*)d_in[9];
    const float* g1 = (const float*)d_in[10];
    const float* s1 = (const float*)d_in[11];
    const float* g2 = (const float*)d_in[12];
    const float* s2 = (const float*)d_in[13];

    char* ws = (char*)d_ws;
    size_t off = 0;
    auto alloc = [&](size_t bytes) -> char* {
        char* p = ws + off;
        off = (off + bytes + 255) & ~(size_t)255;
        return p;
    };

    const size_t DD = (size_t)Dn * Dn, DF = (size_t)Dn * FF, MD = (size_t)MS * Dn, MF = (size_t)MS * FF;
    u16t* wqT  = (u16t*)alloc(DD * 2);   // all weights stored transposed [N][K]
    u16t* wkT  = (u16t*)alloc(DD * 2);
    u16t* wvT  = (u16t*)alloc(DD * 2);
    u16t* woT  = (u16t*)alloc(DD * 2);
    u16t* w1T  = (u16t*)alloc(DF * 2);
    u16t* w2T  = (u16t*)alloc(DF * 2);
    u16t* hB   = (u16t*)alloc(MD * 2);
    u16t* qB   = (u16t*)alloc(MD * 2);
    u16t* kB   = (u16t*)alloc(MD * 2);
    u16t* vTB  = (u16t*)alloc(MD * 2);   // V stored transposed [D][B*S]
    u16t* ctxB = (u16t*)alloc(MD * 2);
    u16t* h2B  = (u16t*)alloc(MD * 2);
    u16t* ffnB = (u16t*)alloc(MF * 2);
    float* x2  = (float*)alloc(MD * 4);

    // weight conversion (f32 -> bf16, transposed)
    dim3 tDD(Dn / 32, Dn / 32);          // (64, 64)
    conv_bf16_t<<<tDD, 256, 0, stream>>>(Wq, wqT, Dn, Dn);
    conv_bf16_t<<<tDD, 256, 0, stream>>>(Wk, wkT, Dn, Dn);
    conv_bf16_t<<<tDD, 256, 0, stream>>>(Wv, wvT, Dn, Dn);
    conv_bf16_t<<<tDD, 256, 0, stream>>>(Wo, woT, Dn, Dn);
    conv_bf16_t<<<dim3(FF / 32, Dn / 32), 256, 0, stream>>>(W1, w1T, Dn, FF);
    conv_bf16_t<<<dim3(Dn / 32, FF / 32), 256, 0, stream>>>(W2, w2T, FF, Dn);

    // attention sublayer
    layernorm_bf16<<<MS, 256, 0, stream>>>(x, g1, s1, hB, Dn);
    dim3 gD(Dn / 128, MS / 128);   // (16, 32)
    gemm_bf16<<<gD, 256, 0, stream>>>(hB, wqT, nullptr, nullptr, nullptr, qB, nullptr, MS, Dn, Dn, 0);
    gemm_bf16<<<gD, 256, 0, stream>>>(hB, wkT, nullptr, nullptr, nullptr, kB, nullptr, MS, Dn, Dn, 0);
    gemm_bf16<<<gD, 256, 0, stream>>>(hB, wvT, nullptr, nullptr, nullptr, nullptr, vTB, MS, Dn, Dn, 0);
    attention_flash<<<Bn * NH_ * (Sn / 16), 32, 0, stream>>>(qB, kB, vTB, ctxB);
    gemm_bf16<<<gD, 256, 0, stream>>>(ctxB, woT, bo, x, x2, nullptr, nullptr, MS, Dn, Dn, 0);

    // FFN sublayer
    layernorm_bf16<<<MS, 256, 0, stream>>>(x2, g2, s2, h2B, Dn);
    dim3 gF(FF / 128, MS / 128);   // (64, 32)
    gemm_bf16<<<gF, 256, 0, stream>>>(h2B, w1T, b1, nullptr, nullptr, ffnB, nullptr, MS, FF, Dn, 1);
    gemm_bf16<<<gD, 256, 0, stream>>>(ffnB, w2T, b2, x2, (float*)d_out, nullptr, nullptr, MS, Dn, FF, 0);
}
